// PhysicsInformedAttention_30236569764097
// MI455X (gfx1250) — matrix operand order repaired and checked by reading, not compiled, not run on hardware
//
#include <hip/hip_runtime.h>
#include <hip/hip_bf16.h>

#define BB 2
#define SEQ 2048
#define HID 1024
#define NH 16
#define DH 64
#define SAUG 2051          // 2048 + 3 physics tokens
#define SP 2080            // padded to multiple of 32
#define MTOT (BB * SP)     // 4160 rows
#define LN_EPS 1e-5f

typedef __attribute__((ext_vector_type(16))) _Float16 v16h;
typedef __attribute__((ext_vector_type(8)))  _Float16 v8h;
typedef __attribute__((ext_vector_type(8)))  float    v8f;

// ---------------------------------------------------------------------------
// CDNA5 async copy: global -> LDS, tracked by ASYNCcnt (ISA 15.18.3 op 98).
// offset applies to BOTH the LDS and global address (ISA 08 pseudocode).
// ---------------------------------------------------------------------------
template <int OFF>
__device__ __forceinline__ void async_load_b128(unsigned lds_addr, const void* gaddr) {
    asm volatile("global_load_async_to_lds_b128 %0, %1, off offset:%2"
                 :: "v"(lds_addr), "v"(gaddr), "i"(OFF) : "memory");
}
__device__ __forceinline__ void wait_async_le5() {
    asm volatile("s_wait_asynccnt 0x5" ::: "memory");
}
__device__ __forceinline__ void wait_async_0() {
    asm volatile("s_wait_asynccnt 0x0" ::: "memory");
}

// ---------------------------------------------------------------------------
// f32 [k][n] -> f16 transposed [n][k] weight conversion (once, off hot path)
// ---------------------------------------------------------------------------
__global__ void cvt_wt_kernel(const float* __restrict__ in, _Float16* __restrict__ out) {
    size_t i = (size_t)blockIdx.x * blockDim.x + threadIdx.x;
    if (i >= (size_t)HID * HID) return;
    int k = (int)(i >> 10);
    int n = (int)(i & (HID - 1));
    out[(size_t)n * HID + k] = (_Float16)in[i];
}

// ---------------------------------------------------------------------------
// Build augmented (features ++ physics tokens, zero pad) sequence in f16.
// ---------------------------------------------------------------------------
__global__ void build_aug_kernel(const float* __restrict__ feat,
                                 const float* __restrict__ e_emb,
                                 const float* __restrict__ m_emb,
                                 const float* __restrict__ p_emb,
                                 _Float16* __restrict__ out) {
    size_t idx = (size_t)blockIdx.x * blockDim.x + threadIdx.x;
    if (idx >= (size_t)BB * SP * HID) return;
    int j = (int)(idx & (HID - 1));
    size_t bs = idx >> 10;
    int s = (int)(bs % SP);
    int b = (int)(bs / SP);
    float v;
    if (s < SEQ)            v = feat[((size_t)b * SEQ + s) * HID + j];
    else if (s == SEQ)      v = e_emb[j];
    else if (s == SEQ + 1)  v = m_emb[j];
    else if (s == SEQ + 2)  v = p_emb[j];
    else                    v = 0.f;
    out[idx] = (_Float16)v;
}

// ---------------------------------------------------------------------------
// WMMA GEMM: Y[M=MTOT, N=HID] = X[M, K=HID](f16) @ W[K, N] + bias,
// with W pre-transposed as Wt[n][k] (f16).
// Block: 256 threads = 8 waves as 2(M) x 4(N); block tile 64 x 256; K step 32.
// Each wave: 2 A frags x 4 B frags -> 8 WMMAs per K step (32x64 output).
// Staging: async global->LDS (ASYNCcnt) with double-buffered tiles so the
// next tile's fetch fully overlaps the current tile's 8 WMMAs.
// mode 0: f16 row-major [m][n]            (Q, K, attention output buffer)
// mode 1: f32 row-major [m][n]            (O projection)
// mode 2: f16 transposed [b][h][d][s]     (V^T for attention's P.V)
// ---------------------------------------------------------------------------
__global__ __launch_bounds__(256) void gemm_wmma_kernel(
        const _Float16* __restrict__ X, const _Float16* __restrict__ Wt,
        const float* __restrict__ bias, void* __restrict__ out, int mode) {
    int mt = blockIdx.x >> 2;       // 0..64
    int nt = blockIdx.x & 3;        // 0..3
    int t = threadIdx.x;
    int wave = t >> 5;              // 0..7
    int lane = t & 31;
    int hlane = lane & 15;
    bool hi = lane >= 16;

    int m0 = mt * 64;
    int n0 = nt * 256;
    int wrow = (wave >> 2) * 32;    // 0 / 32
    int wcol = (wave & 3) * 64;     // 0,64,128,192

    __shared__ __align__(16) _Float16 As[2][64 * 32];   // [m][k] double-buffered
    __shared__ __align__(16) _Float16 Bs[2][256 * 32];  // [n][k] double-buffered

    // Per-thread staging addresses: 8 halves of A (1 x b128), 32 halves of B (4 x b128)
    int aidx = t * 8;
    int ar = aidx >> 5, ac = aidx & 31;
    const _Float16* gA = &X[(size_t)(m0 + ar) * HID + ac];   // + k0
    const _Float16* gB = &Wt[(size_t)(n0 + t) * HID];        // + k0
    unsigned ldsA[2] = { (unsigned)(size_t)&As[0][aidx],  (unsigned)(size_t)&As[1][aidx] };
    unsigned ldsB[2] = { (unsigned)(size_t)&Bs[0][t * 32], (unsigned)(size_t)&Bs[1][t * 32] };

    v8f acc[2][4];
    #pragma unroll
    for (int i = 0; i < 2; ++i)
        #pragma unroll
        for (int j = 0; j < 4; ++j) acc[i][j] = v8f{};

    const int NIT = HID / 32;       // 32 K-steps

    // Prologue: async-fetch tile 0 into buffer 0 (5 async instructions)
    async_load_b128<0>(ldsA[0], gA);
    async_load_b128<0>(ldsB[0], gB);
    async_load_b128<16>(ldsB[0], gB);
    async_load_b128<32>(ldsB[0], gB);
    async_load_b128<48>(ldsB[0], gB);

    for (int it = 0; it < NIT; ++it) {
        int cur = it & 1;
        if (it + 1 < NIT) {
            // Fetch next tile into the other buffer while we compute this one
            int k1 = (it + 1) * 32;
            int nxt = cur ^ 1;
            async_load_b128<0>(ldsA[nxt], gA + k1);
            async_load_b128<0>(ldsB[nxt], gB + k1);
            async_load_b128<16>(ldsB[nxt], gB + k1);
            async_load_b128<32>(ldsB[nxt], gB + k1);
            async_load_b128<48>(ldsB[nxt], gB + k1);
            wait_async_le5();       // tile `it` complete (in-order), tile it+1 in flight
        } else {
            wait_async_0();
        }
        __syncthreads();

        // A fragments (lanes<16: K=0..15; lanes>=16: K=16..31)
        v16h a[2];
        #pragma unroll
        for (int i = 0; i < 2; ++i) {
            int ks = hi ? 16 : 0;
            const _Float16* ap = &As[cur][(wrow + i * 16 + hlane) * 32];
            v8h lo = *(const v8h*)(ap + ks);
            v8h hh = *(const v8h*)(ap + ks + 8);
            #pragma unroll
            for (int e = 0; e < 8; ++e) { a[i][e] = lo[e]; a[i][e + 8] = hh[e]; }
        }
        // B fragments (lanes<16: K=0..15; lanes>=16: K=16..31)
        v16h bf[4];
        #pragma unroll
        for (int j = 0; j < 4; ++j) {
            const _Float16* bp = &Bs[cur][(wcol + j * 16 + hlane) * 32 + (hi ? 16 : 0)];
            v8h x0 = *(const v8h*)bp, x1 = *(const v8h*)(bp + 8);
            #pragma unroll
            for (int e = 0; e < 8; ++e) { bf[j][e] = x0[e]; bf[j][e + 8] = x1[e]; }
        }
        #pragma unroll
        for (int i = 0; i < 2; ++i)
            #pragma unroll
            for (int j = 0; j < 4; ++j)
                acc[i][j] = __builtin_amdgcn_wmma_f32_16x16x32_f16(
                    false, a[i], false, bf[j], (short)0, acc[i][j], false, false);
        __syncthreads();            // all waves done with buf[cur] before it is refilled
    }

    // Epilogue
    #pragma unroll
    for (int j = 0; j < 4; ++j) {
        int n = n0 + wcol + j * 16 + hlane;
        float bn = bias[n];
        #pragma unroll
        for (int i = 0; i < 2; ++i) {
            int mbase = m0 + wrow + i * 16 + (hi ? 8 : 0);
            if (mode == 2) {
                // V^T: [b][h][d][s]; 8 consecutive rows = 8 consecutive s -> b128
                int b = mbase / SP, s = mbase % SP;   // 8-run never crosses b
                int h = n >> 6, d = n & 63;
                v8h pack;
                #pragma unroll
                for (int r = 0; r < 8; ++r) pack[r] = (_Float16)(acc[i][j][r] + bn);
                _Float16* o = (_Float16*)out;
                *(v8h*)&o[((size_t)(b * NH + h) * DH + d) * SP + s] = pack;
            } else if (mode == 1) {
                float* o = (float*)out;
                #pragma unroll
                for (int r = 0; r < 8; ++r)
                    o[(size_t)(mbase + r) * HID + n] = acc[i][j][r] + bn;
            } else {
                _Float16* o = (_Float16*)out;
                #pragma unroll
                for (int r = 0; r < 8; ++r)
                    o[(size_t)(mbase + r) * HID + n] = (_Float16)(acc[i][j][r] + bn);
            }
        }
    }
}

// ---------------------------------------------------------------------------
// Flash-style attention, one wave per 16-row q tile per (b,h).
// Q,K: [b*SP+s][HID] f16 (head h at column h*DH).  Vt: [b][h][d][s] f16.
// Physics biases are row-constant -> softmax invariant -> dropped.
// att out: [b*SP+s][h*DH+d] f16 (GEMM-ready).
// ---------------------------------------------------------------------------
__global__ __launch_bounds__(32) void attn_wmma_kernel(
        const _Float16* __restrict__ Q, const _Float16* __restrict__ K,
        const _Float16* __restrict__ Vt, _Float16* __restrict__ att) {
    const int QT = SP / 16;                 // 130 q tiles
    int qtile = blockIdx.x % QT;
    int bh = blockIdx.x / QT;
    int b = bh / NH, h = bh % NH;
    int lane = threadIdx.x;
    int hlane = lane & 15;
    bool hi = lane >= 16;
    int q0 = qtile * 16;

    const _Float16* Qb = Q + ((size_t)(b * SP)) * HID + h * DH;
    const _Float16* Kb = K + ((size_t)(b * SP)) * HID + h * DH;
    const _Float16* Vb = Vt + (size_t)bh * DH * SP;

    // Persistent Q A-fragments for the two 32-wide d chunks
    v16h qa[2];
    #pragma unroll
    for (int c = 0; c < 2; ++c) {
        int ks = c * 32 + (hi ? 16 : 0);
        const _Float16* qp = Qb + (size_t)(q0 + hlane) * HID;
        v8h lo = *(const v8h*)(qp + ks);
        v8h hh = *(const v8h*)(qp + ks + 8);
        #pragma unroll
        for (int e = 0; e < 8; ++e) { qa[c][e] = lo[e]; qa[c][e + 8] = hh[e]; }
    }

    __shared__ __align__(16) _Float16 Plds[16 * 32]; // probability bounce buffer

    float m_run[8], l_run[8];
    v8f O[4] = {v8f{}, v8f{}, v8f{}, v8f{}};
    #pragma unroll
    for (int r = 0; r < 8; ++r) { m_run[r] = -1e30f; l_run[r] = 0.f; }

    for (int k0 = 0; k0 < SP; k0 += 32) {
        // Scores S[q,k] = sum_d Q[q,d] K[k,d]; B frags direct from global
        v8f S0 = {}, S1 = {};
        #pragma unroll
        for (int c = 0; c < 2; ++c) {
            int ds = c * 32 + (hi ? 16 : 0);
            const _Float16* kp0 = Kb + (size_t)(k0 + hlane) * HID + ds;
            const _Float16* kp1 = Kb + (size_t)(k0 + 16 + hlane) * HID + ds;
            v16h b0, b1;
            v8h x0 = *(const v8h*)kp0, x1 = *(const v8h*)(kp0 + 8);
            v8h y0 = *(const v8h*)kp1, y1 = *(const v8h*)(kp1 + 8);
            #pragma unroll
            for (int e = 0; e < 8; ++e) {
                b0[e] = x0[e]; b0[e + 8] = x1[e];
                b1[e] = y0[e]; b1[e + 8] = y1[e];
            }
            S0 = __builtin_amdgcn_wmma_f32_16x16x32_f16(false, qa[c], false, b0, (short)0, S0, false, false);
            S1 = __builtin_amdgcn_wmma_f32_16x16x32_f16(false, qa[c], false, b1, (short)0, S1, false, false);
        }
        if (k0 + 32 < SP) {  // prefetch next K rows -> global_prefetch_b8
            __builtin_prefetch(Kb + (size_t)(k0 + 32 + lane) * HID, 0, 0);
        }

        // Scale 1/sqrt(64) and mask padded k columns
        const float scale = 0.125f;
        float cm0 = (k0 + hlane) < SAUG ? 0.f : -1e30f;
        float cm1 = (k0 + 16 + hlane) < SAUG ? 0.f : -1e30f;
        #pragma unroll
        for (int r = 0; r < 8; ++r) {
            S0[r] = S0[r] * scale + cm0;
            S1[r] = S1[r] * scale + cm1;
        }

        // Online softmax: row reductions within each 16-lane half
        #pragma unroll
        for (int r = 0; r < 8; ++r) {
            float mx = fmaxf(S0[r], S1[r]);
            #pragma unroll
            for (int off = 1; off < 16; off <<= 1) mx = fmaxf(mx, __shfl_xor(mx, off, 32));
            float mnew = fmaxf(m_run[r], mx);
            float corr = __expf(m_run[r] - mnew);
            m_run[r] = mnew;
            float p0 = __expf(S0[r] - mnew);
            float p1 = __expf(S1[r] - mnew);
            S0[r] = p0; S1[r] = p1;
            float rs = p0 + p1;
            #pragma unroll
            for (int off = 1; off < 16; off <<= 1) rs += __shfl_xor(rs, off, 32);
            l_run[r] = l_run[r] * corr + rs;
            #pragma unroll
            for (int c = 0; c < 4; ++c) O[c][r] *= corr;
        }

        // Bounce P through LDS: C-frag layout -> A-frag layout (f16)
        #pragma unroll
        for (int r = 0; r < 8; ++r) {
            int row = r + (hi ? 8 : 0);
            Plds[row * 32 + hlane]      = (_Float16)S0[r];
            Plds[row * 32 + 16 + hlane] = (_Float16)S1[r];
        }
        __syncthreads();

        v16h pa;
        {
            int ks = hi ? 16 : 0;
            const _Float16* pp = &Plds[hlane * 32];
            v8h lo = *(const v8h*)(pp + ks);
            v8h hh = *(const v8h*)(pp + ks + 8);
            #pragma unroll
            for (int e = 0; e < 8; ++e) { pa[e] = lo[e]; pa[e + 8] = hh[e]; }
        }
        // O[q,d] += P[q,kk] V[kk,d]; B frags direct from transposed Vt
        #pragma unroll
        for (int c = 0; c < 4; ++c) {
            const _Float16* vp = Vb + (size_t)(c * 16 + hlane) * SP + k0 + (hi ? 16 : 0);
            v8h x0 = *(const v8h*)vp, x1 = *(const v8h*)(vp + 8);
            v16h vb;
            #pragma unroll
            for (int e = 0; e < 8; ++e) { vb[e] = x0[e]; vb[e + 8] = x1[e]; }
            O[c] = __builtin_amdgcn_wmma_f32_16x16x32_f16(false, pa, false, vb, (short)0, O[c], false, false);
        }
        __syncthreads();
    }

    // Normalize and store to [b*SP+s][h*DH+d]
    #pragma unroll
    for (int r = 0; r < 8; ++r) {
        int row = r + (hi ? 8 : 0);
        int s = q0 + row;
        float inv = 1.f / l_run[r];
        _Float16* op = att + ((size_t)(b * SP + s)) * HID + h * DH;
        #pragma unroll
        for (int c = 0; c < 4; ++c) op[c * 16 + hlane] = (_Float16)(O[c][r] * inv);
    }
}

// ---------------------------------------------------------------------------
// Residual + LayerNorm: out[b,s,:] = LN(resid[b,s,:] + y[b,s(padded),:])
// ---------------------------------------------------------------------------
__global__ __launch_bounds__(256) void ln_kernel(
        const float* __restrict__ y, const float* __restrict__ resid,
        const float* __restrict__ gamma, const float* __restrict__ beta,
        float* __restrict__ out) {
    int row = blockIdx.x;               // 0 .. B*SEQ-1
    int b = row / SEQ, s = row % SEQ;
    const float* yr = y + ((size_t)b * SP + s) * HID;
    const float* rr = resid + (size_t)row * HID;
    __shared__ float red[256];
    int t = threadIdx.x;
    float x[4];
    float sum = 0.f;
    #pragma unroll
    for (int i = 0; i < 4; ++i) { x[i] = yr[t + i * 256] + rr[t + i * 256]; sum += x[i]; }
    red[t] = sum; __syncthreads();
    for (int o = 128; o > 0; o >>= 1) { if (t < o) red[t] += red[t + o]; __syncthreads(); }
    float mu = red[0] * (1.f / HID);
    __syncthreads();
    float vs = 0.f;
    #pragma unroll
    for (int i = 0; i < 4; ++i) { float d = x[i] - mu; vs += d * d; }
    red[t] = vs; __syncthreads();
    for (int o = 128; o > 0; o >>= 1) { if (t < o) red[t] += red[t + o]; __syncthreads(); }
    float rinv = rsqrtf(red[0] * (1.f / HID) + LN_EPS);
    float* orow = out + (size_t)row * HID;
    #pragma unroll
    for (int i = 0; i < 4; ++i) {
        int j = t + i * 256;
        orow[j] = (x[i] - mu) * rinv * gamma[j] + beta[j];
    }
}

// ---------------------------------------------------------------------------
extern "C" void kernel_launch(void* const* d_in, const int* in_sizes, int n_in,
                              void* d_out, int out_size, void* d_ws, size_t ws_size,
                              hipStream_t stream) {
    const float* cnn = (const float*)d_in[0];
    const float* llm = (const float*)d_in[1];
    const float* Wq = (const float*)d_in[2];  const float* bq = (const float*)d_in[3];
    const float* Wk = (const float*)d_in[4];  const float* bk = (const float*)d_in[5];
    const float* Wv = (const float*)d_in[6];  const float* bv = (const float*)d_in[7];
    const float* Wo = (const float*)d_in[8];  const float* bo = (const float*)d_in[9];
    const float* ee = (const float*)d_in[10];
    const float* me = (const float*)d_in[11];
    const float* pe = (const float*)d_in[12];
    const float* gamma = (const float*)d_in[13];
    const float* beta  = (const float*)d_in[14];
    float* out = (float*)d_out;

    char* p = (char*)d_ws;
    auto alloc = [&](size_t bytes) -> void* {
        void* r = (void*)p;
        p += (bytes + 255) & ~(size_t)255;
        return r;
    };

    const size_t nAug = (size_t)BB * SP * HID;
    const size_t nW   = (size_t)HID * HID;

    _Float16* augC = (_Float16*)alloc(nAug * 2);
    _Float16* augL = (_Float16*)alloc(nAug * 2);
    _Float16* WqT = (_Float16*)alloc(nW * 2);
    _Float16* WkT = (_Float16*)alloc(nW * 2);
    _Float16* WvT = (_Float16*)alloc(nW * 2);
    _Float16* WoT = (_Float16*)alloc(nW * 2);
    _Float16* Qc = (_Float16*)alloc(nAug * 2);
    _Float16* Kc = (_Float16*)alloc(nAug * 2);
    _Float16* VcT = (_Float16*)alloc(nAug * 2);
    _Float16* Ql = (_Float16*)alloc(nAug * 2);
    _Float16* Kl = (_Float16*)alloc(nAug * 2);
    _Float16* VlT = (_Float16*)alloc(nAug * 2);
    _Float16* attC = (_Float16*)alloc(nAug * 2);
    _Float16* attL = (_Float16*)alloc(nAug * 2);
    float* ybufC = (float*)alloc(nAug * 4);
    float* ybufL = (float*)alloc(nAug * 4);

    // 1. Weight conversion (+ transpose) and augmented sequence build
    {
        int g = (int)((nW + 255) / 256);
        cvt_wt_kernel<<<g, 256, 0, stream>>>(Wq, WqT);
        cvt_wt_kernel<<<g, 256, 0, stream>>>(Wk, WkT);
        cvt_wt_kernel<<<g, 256, 0, stream>>>(Wv, WvT);
        cvt_wt_kernel<<<g, 256, 0, stream>>>(Wo, WoT);
    }
    {
        int g = (int)((nAug + 255) / 256);
        build_aug_kernel<<<g, 256, 0, stream>>>(cnn, ee, me, pe, augC);
        build_aug_kernel<<<g, 256, 0, stream>>>(llm, ee, me, pe, augL);
    }

    // 2. Q/K/V projections (WMMA GEMM, async double-buffered staging)
    dim3 ggrid((MTOT / 64) * (HID / 256));   // 65 * 4 = 260
    gemm_wmma_kernel<<<ggrid, 256, 0, stream>>>(augC, WqT, bq, Qc, 0);
    gemm_wmma_kernel<<<ggrid, 256, 0, stream>>>(augC, WkT, bk, Kc, 0);
    gemm_wmma_kernel<<<ggrid, 256, 0, stream>>>(augC, WvT, bv, VcT, 2);
    gemm_wmma_kernel<<<ggrid, 256, 0, stream>>>(augL, WqT, bq, Ql, 0);
    gemm_wmma_kernel<<<ggrid, 256, 0, stream>>>(augL, WkT, bk, Kl, 0);
    gemm_wmma_kernel<<<ggrid, 256, 0, stream>>>(augL, WvT, bv, VlT, 2);

    // 3. Attention (flash-style, WMMA)
    {
        int g = BB * NH * (SP / 16);         // 4160
        attn_wmma_kernel<<<g, 32, 0, stream>>>(Qc, Kl, VlT, attC);  // cnn queries llm
        attn_wmma_kernel<<<g, 32, 0, stream>>>(Ql, Kc, VcT, attL);  // llm queries cnn
    }

    // 4. Output projection (WMMA GEMM, f32 output)
    gemm_wmma_kernel<<<ggrid, 256, 0, stream>>>(attC, WoT, bo, ybufC, 1);
    gemm_wmma_kernel<<<ggrid, 256, 0, stream>>>(attL, WoT, bo, ybufL, 1);

    // 5. Residual + LayerNorm into the two concatenated outputs
    {
        int g = BB * SEQ;
        ln_kernel<<<g, 256, 0, stream>>>(ybufC, cnn, gamma, beta, out);
        ln_kernel<<<g, 256, 0, stream>>>(ybufL, llm, gamma, beta, out + (size_t)BB * SEQ * HID);
    }
}
